// SelfAttention_59279138619544
// MI455X (gfx1250) — matrix operand order repaired and checked by reading, not compiled, not run on hardware
//
#include <hip/hip_runtime.h>
#include <hip/hip_bf16.h>

typedef __attribute__((ext_vector_type(16))) _Float16 v16h;
typedef __attribute__((ext_vector_type(8)))  _Float16 v8h;
typedef __attribute__((ext_vector_type(2)))  _Float16 v2h;
typedef __attribute__((ext_vector_type(8)))  float    v8f;

#define EMBED 1024
#define SEQ   2048
#define BATCH 2
#define NHEAD 16
#define HDIM  64
#define MTOT  (BATCH * SEQ)   // 4096 rows

union F16x16 { v16h v; v8h h[2]; };

__device__ __forceinline__ v8f wmma_f16(const v16h& a, const v16h& b, const v8f& c) {
    return __builtin_amdgcn_wmma_f32_16x16x32_f16(
        /*neg_a=*/false, a, /*neg_b=*/false, b,
        /*c_mod=*/(short)0, c, /*reuse_a=*/false, /*reuse_b=*/false);
}

// ---------------------------------------------------------------------------
// Kernel 1: fused QKV projection.  Y = X @ W + b (f16 out).
//   Q,K stored [B,H,S,D];  V stored transposed [B,H,D,S] so the attention
//   P@V B-fragments are contiguous b128 loads.  Q pre-scaled by 1/sqrt(D).
// Block 256 thr (8 waves, 4x2). Tile 128x64, K-step 32, wave tile 32x32.
// ---------------------------------------------------------------------------
__global__ __launch_bounds__(256)
void qkv_gemm(const float* __restrict__ X,
              const float* __restrict__ Wq, const float* __restrict__ bq,
              const float* __restrict__ Wk, const float* __restrict__ bk,
              const float* __restrict__ Wv, const float* __restrict__ bv,
              _Float16* __restrict__ Q, _Float16* __restrict__ K,
              _Float16* __restrict__ Vt)
{
    const float* W; const float* bias; _Float16* out; float scale;
    if (blockIdx.z == 0)      { W = Wq; bias = bq; out = Q;  scale = 0.125f; }
    else if (blockIdx.z == 1) { W = Wk; bias = bk; out = K;  scale = 1.0f;   }
    else                      { W = Wv; bias = bv; out = Vt; scale = 1.0f;   }
    const bool isV = (blockIdx.z == 2);

    __shared__ __align__(16) _Float16 As [128][40];  // 128 x 32 (+pad), K-contig
    __shared__ __align__(16) _Float16 BsT[ 64][40];  // transposed: [n][k]

    const int tid  = threadIdx.x;
    const int lane = tid & 31;
    const int wave = tid >> 5;
    const int wr   = wave >> 1;       // 0..3  -> 32-row strip
    const int wc   = wave & 1;        // 0..1  -> 32-col strip
    const int grp  = lane >> 4;       // half-wave id
    const int l16  = lane & 15;

    const int mbase = blockIdx.y * 128;
    const int nbase = blockIdx.x * 64;

    v8f acc[2][2] = {{{}, {}}, {{}, {}}};

    for (int kb = 0; kb < EMBED; kb += 32) {
        {   // A tile: 128x32 f32 -> f16, 16 elems/thread, vector LDS stores
            int r  = tid >> 1;
            int c0 = (tid & 1) * 16;
            const float* src = X + (size_t)(mbase + r) * EMBED + kb + c0;
            __builtin_prefetch(src + 32, 0, 1);     // next K-tile
            v8h t0, t1;
            #pragma unroll
            for (int j = 0; j < 8; ++j) { t0[j] = (_Float16)src[j]; t1[j] = (_Float16)src[8 + j]; }
            *(v8h*)&As[r][c0]     = t0;
            *(v8h*)&As[r][c0 + 8] = t1;
        }
        {   // B tile: 32x64 f32, stored transposed (K-contiguous per column)
            int n0 = (tid & 15) * 4;
            int k0 = (tid >> 4) * 2;
            const float* s0 = W + (size_t)(kb + k0) * EMBED + nbase + n0;
            const float* s1 = s0 + EMBED;
            __builtin_prefetch(s0 + 32 * EMBED, 0, 1);
            #pragma unroll
            for (int j = 0; j < 4; ++j) {
                v2h p; p[0] = (_Float16)s0[j]; p[1] = (_Float16)s1[j];
                *(v2h*)&BsT[n0 + j][k0] = p;
            }
        }
        __syncthreads();

        F16x16 a0, a1, b0, b1;
        {
            int r0 = wr * 32 + l16;
            a0.h[0] = *(const v8h*)&As[r0][grp * 16];
            a0.h[1] = *(const v8h*)&As[r0][grp * 16 + 8];
            a1.h[0] = *(const v8h*)&As[r0 + 16][grp * 16];
            a1.h[1] = *(const v8h*)&As[r0 + 16][grp * 16 + 8];
        }
        {
            int c0 = wc * 32 + l16;
            b0.h[0] = *(const v8h*)&BsT[c0][grp * 16];
            b0.h[1] = *(const v8h*)&BsT[c0][grp * 16 + 8];
            b1.h[0] = *(const v8h*)&BsT[c0 + 16][grp * 16];
            b1.h[1] = *(const v8h*)&BsT[c0 + 16][grp * 16 + 8];
        }
        acc[0][0] = wmma_f16(a0.v, b0.v, acc[0][0]);
        acc[0][1] = wmma_f16(a0.v, b1.v, acc[0][1]);
        acc[1][0] = wmma_f16(a1.v, b0.v, acc[1][0]);
        acc[1][1] = wmma_f16(a1.v, b1.v, acc[1][1]);
        __syncthreads();
    }

    // Epilogue: +bias, *scale, scatter
    #pragma unroll
    for (int i = 0; i < 2; ++i) {
        #pragma unroll
        for (int j = 0; j < 2; ++j) {
            v8f a = acc[i][j];
            int n = nbase + wc * 32 + j * 16 + l16;
            int h = n >> 6, d = n & 63;
            float bb = bias[n];
            int m0 = mbase + wr * 32 + i * 16 + grp * 8;   // first C-layout row
            int bi = m0 >> 11;
            int s0 = m0 & (SEQ - 1);
            if (!isV) {
                _Float16* dst = out + (((size_t)bi * NHEAD + h) * SEQ + s0) * HDIM + d;
                #pragma unroll
                for (int r = 0; r < 8; ++r)
                    dst[(size_t)r * HDIM] = (_Float16)((a[r] + bb) * scale);
            } else {
                v8h t;
                #pragma unroll
                for (int r = 0; r < 8; ++r) t[r] = (_Float16)(a[r] + bb);
                *(v8h*)(out + (((size_t)bi * NHEAD + h) * HDIM + d) * SEQ + s0) = t;
            }
        }
    }
}

// ---------------------------------------------------------------------------
// Kernel 2: flash attention.  Block 128 thr (4 waves), one (b,h) x 64 q-rows.
// Q,K in [B,H,S,D] f16; V in [B,H,D,S] f16.  All fragments = b128 loads.
// ---------------------------------------------------------------------------
__global__ __launch_bounds__(128)
void attn_kernel(const _Float16* __restrict__ Q, const _Float16* __restrict__ K,
                 const _Float16* __restrict__ Vt, _Float16* __restrict__ Oattn)
{
    __shared__ __align__(16) _Float16 Pl[4][16][40];   // per-wave P bounce (16x32)

    const int tid  = threadIdx.x;
    const int lane = tid & 31;
    const int wave = tid >> 5;
    const int grp  = lane >> 4;
    const int l16  = lane & 15;

    const int bh    = blockIdx.x;                 // b*NHEAD + h
    const int qbase = blockIdx.y * 64 + wave * 16;

    const _Float16* Qp = Q  + (size_t)bh * SEQ * HDIM;
    const _Float16* Kp = K  + (size_t)bh * SEQ * HDIM;
    const _Float16* Vp = Vt + (size_t)bh * HDIM * SEQ;

    // Q fragments: 16 rows x 64 d  ->  two k=32 fragments, all b128 loads
    F16x16 aq0, aq1;
    {
        const v8h* qv = (const v8h*)(Qp + (size_t)(qbase + l16) * HDIM);
        aq0.h[0] = qv[grp * 2];     aq0.h[1] = qv[grp * 2 + 1];
        aq1.h[0] = qv[4 + grp * 2]; aq1.h[1] = qv[4 + grp * 2 + 1];
    }

    float mrow[8], lrow[8];
    v8f o[4] = {{}, {}, {}, {}};
    #pragma unroll
    for (int r = 0; r < 8; ++r) { mrow[r] = -1e30f; lrow[r] = 0.0f; }

    for (int kb = 0; kb < SEQ; kb += 32) {
        // ---- scores: two 16x16 tiles ----
        v8f s0 = {}, s1 = {};
        {
            const v8h* kv0 = (const v8h*)(Kp + (size_t)(kb + l16) * HDIM);
            const v8h* kv1 = (const v8h*)(Kp + (size_t)(kb + 16 + l16) * HDIM);
            F16x16 b0a, b0b, b1a, b1b;   // B = K^T: element e -> d = base + e
            b0a.h[0] = kv0[grp * 2];     b0a.h[1] = kv0[grp * 2 + 1];
            b0b.h[0] = kv0[4 + grp * 2]; b0b.h[1] = kv0[4 + grp * 2 + 1];
            b1a.h[0] = kv1[grp * 2];     b1a.h[1] = kv1[grp * 2 + 1];
            b1b.h[0] = kv1[4 + grp * 2]; b1b.h[1] = kv1[4 + grp * 2 + 1];
            s0 = wmma_f16(aq0.v, b0a.v, s0); s0 = wmma_f16(aq1.v, b0b.v, s0);
            s1 = wmma_f16(aq0.v, b1a.v, s1); s1 = wmma_f16(aq1.v, b1b.v, s1);
        }

        // ---- online softmax (row stats within 16-lane half) ----
        float p0[8], p1[8];
        #pragma unroll
        for (int r = 0; r < 8; ++r) {
            float v = fmaxf(s0[r], s1[r]);
            v = fmaxf(v, __shfl_xor(v, 1, 32));
            v = fmaxf(v, __shfl_xor(v, 2, 32));
            v = fmaxf(v, __shfl_xor(v, 4, 32));
            v = fmaxf(v, __shfl_xor(v, 8, 32));
            float nm    = fmaxf(mrow[r], v);
            float alpha = __expf(mrow[r] - nm);
            mrow[r] = nm;
            float e0 = __expf(s0[r] - nm);
            float e1 = __expf(s1[r] - nm);
            float rs = e0 + e1;
            rs += __shfl_xor(rs, 1, 32);
            rs += __shfl_xor(rs, 2, 32);
            rs += __shfl_xor(rs, 4, 32);
            rs += __shfl_xor(rs, 8, 32);
            lrow[r] = lrow[r] * alpha + rs;
            o[0][r] *= alpha; o[1][r] *= alpha; o[2][r] *= alpha; o[3][r] *= alpha;
            p0[r] = e0; p1[r] = e1;
        }

        // ---- C-layout -> A-layout for P via per-wave LDS bounce ----
        __syncthreads();
        #pragma unroll
        for (int r = 0; r < 8; ++r) {
            int row = grp * 8 + r;
            Pl[wave][row][l16]      = (_Float16)p0[r];
            Pl[wave][row][16 + l16] = (_Float16)p1[r];
        }
        __syncthreads();
        F16x16 pa;
        pa.h[0] = *(const v8h*)&Pl[wave][l16][grp * 8];
        pa.h[1] = *(const v8h*)&Pl[wave][l16][16 + grp * 8];

        // ---- O += P(16x32) @ V_block(32x64); V^T fragments are contiguous ----
        #pragma unroll
        for (int dt = 0; dt < 4; ++dt) {
            const v8h* vv = (const v8h*)(Vp + (size_t)(dt * 16 + l16) * SEQ + kb);
            F16x16 bv_;
            bv_.h[0] = vv[grp];
            bv_.h[1] = vv[2 + grp];
            o[dt] = wmma_f16(pa.v, bv_.v, o[dt]);
        }
    }

    // ---- finalize: /l, store f16 into [B,S,E] ----
    const int b = bh >> 4, h = bh & 15;
    #pragma unroll
    for (int r = 0; r < 8; ++r) {
        float inv = 1.0f / lrow[r];
        int s = qbase + grp * 8 + r;
        _Float16* dst = Oattn + ((size_t)b * SEQ + s) * EMBED + h * HDIM + l16;
        dst[ 0] = (_Float16)(o[0][r] * inv);
        dst[16] = (_Float16)(o[1][r] * inv);
        dst[32] = (_Float16)(o[2][r] * inv);
        dst[48] = (_Float16)(o[3][r] * inv);
    }
}

// ---------------------------------------------------------------------------
// Kernel 3: output projection.  out = Aattn(f16) @ Wo + bo -> f32.
// A fragments straight from global (row-contiguous b128); Wo via LDS (f32->f16,
// transposed).  Block 256 thr, tile 128x64, wave tile 32x32.
// ---------------------------------------------------------------------------
__global__ __launch_bounds__(256)
void out_proj(const _Float16* __restrict__ A, const float* __restrict__ Wo,
              const float* __restrict__ bo, float* __restrict__ out)
{
    __shared__ __align__(16) _Float16 BsT[64][40];   // transposed: [n][k]

    const int tid  = threadIdx.x;
    const int lane = tid & 31;
    const int wave = tid >> 5;
    const int wr   = wave >> 1;
    const int wc   = wave & 1;
    const int grp  = lane >> 4;
    const int l16  = lane & 15;

    const int mbase = blockIdx.y * 128;
    const int nbase = blockIdx.x * 64;

    v8f acc[2][2] = {{{}, {}}, {{}, {}}};

    for (int kb = 0; kb < EMBED; kb += 32) {
        {   // B tile: 32x64 f32 -> f16 transposed
            int n0 = (tid & 15) * 4;
            int k0 = (tid >> 4) * 2;
            const float* s0 = Wo + (size_t)(kb + k0) * EMBED + nbase + n0;
            const float* s1 = s0 + EMBED;
            __builtin_prefetch(s0 + 32 * EMBED, 0, 1);
            #pragma unroll
            for (int j = 0; j < 4; ++j) {
                v2h p; p[0] = (_Float16)s0[j]; p[1] = (_Float16)s1[j];
                *(v2h*)&BsT[n0 + j][k0] = p;
            }
        }
        __syncthreads();

        F16x16 a0, a1, b0, b1;
        {   // A fragments straight from global memory
            const v8h* av0 = (const v8h*)(A + (size_t)(mbase + wr * 32 + l16) * EMBED + kb);
            const v8h* av1 = (const v8h*)(A + (size_t)(mbase + wr * 32 + 16 + l16) * EMBED + kb);
            a0.h[0] = av0[grp * 2]; a0.h[1] = av0[grp * 2 + 1];
            a1.h[0] = av1[grp * 2]; a1.h[1] = av1[grp * 2 + 1];
        }
        {
            int c0 = wc * 32 + l16;
            b0.h[0] = *(const v8h*)&BsT[c0][grp * 16];
            b0.h[1] = *(const v8h*)&BsT[c0][grp * 16 + 8];
            b1.h[0] = *(const v8h*)&BsT[c0 + 16][grp * 16];
            b1.h[1] = *(const v8h*)&BsT[c0 + 16][grp * 16 + 8];
        }
        acc[0][0] = wmma_f16(a0.v, b0.v, acc[0][0]);
        acc[0][1] = wmma_f16(a0.v, b1.v, acc[0][1]);
        acc[1][0] = wmma_f16(a1.v, b0.v, acc[1][0]);
        acc[1][1] = wmma_f16(a1.v, b1.v, acc[1][1]);
        __syncthreads();
    }

    #pragma unroll
    for (int i = 0; i < 2; ++i) {
        #pragma unroll
        for (int j = 0; j < 2; ++j) {
            v8f a = acc[i][j];
            int n = nbase + wc * 32 + j * 16 + l16;
            float bb = bo[n];
            int m0 = mbase + wr * 32 + i * 16 + grp * 8;
            float* dst = out + (size_t)m0 * EMBED + n;
            #pragma unroll
            for (int r = 0; r < 8; ++r)
                dst[(size_t)r * EMBED] = a[r] + bb;
        }
    }
}

// ---------------------------------------------------------------------------
extern "C" void kernel_launch(void* const* d_in, const int* in_sizes, int n_in,
                              void* d_out, int out_size, void* d_ws, size_t ws_size,
                              hipStream_t stream)
{
    const float* x  = (const float*)d_in[0];
    const float* Wq = (const float*)d_in[1];
    const float* bq = (const float*)d_in[2];
    const float* Wk = (const float*)d_in[3];
    const float* bk = (const float*)d_in[4];
    const float* Wv = (const float*)d_in[5];
    const float* bv = (const float*)d_in[6];
    const float* Wo = (const float*)d_in[7];
    const float* bo = (const float*)d_in[8];

    const size_t perTensor = (size_t)BATCH * NHEAD * SEQ * HDIM;  // 4M f16 = 8MB
    _Float16* Q     = (_Float16*)d_ws;
    _Float16* K     = Q + perTensor;
    _Float16* Vt    = K + perTensor;   // [B,H,D,S]
    _Float16* Aattn = Vt + perTensor;  // [B,S,E] f16

    dim3 g1(EMBED / 64, MTOT / 128, 3);
    qkv_gemm<<<g1, 256, 0, stream>>>(x, Wq, bq, Wk, bk, Wv, bv, Q, K, Vt);

    dim3 g2(BATCH * NHEAD, SEQ / 64);
    attn_kernel<<<g2, 128, 0, stream>>>(Q, K, Vt, Aattn);

    dim3 g3(EMBED / 64, MTOT / 128);
    out_proj<<<g3, 256, 0, stream>>>(Aattn, Wo, bo, (float*)d_out);
}
